// Model_81406810128983
// MI455X (gfx1250) — matrix operand order repaired and checked by reading, not compiled, not run on hardware
//
#include <hip/hip_runtime.h>

typedef __bf16 bf16;
typedef __attribute__((ext_vector_type(16))) __bf16 v16bf;
typedef __attribute__((ext_vector_type(8)))  float  v8f;

union FragU { uint4 q[2]; v16bf v; };

struct Geom {
    int H, W, C;       // input NHWC dims
    int iShift;        // log2(channels per conv group): 8 (prim) or 3 (capsule atoms)
    int PS;            // kernel positions (R*S)
    int S;             // kernel width
    int stride;
    int OW, OHOW;      // output spatial
    int G;             // groups per batch sample (in_dim)
};

// ---------------------------------------------------------------------------
// Weight pack: Bpack[(kblock*N + n)*32 + kk] = W[k][n] (bf16, zero-padded K)
// mode 0: src [K][N] f32 (FC). mode 1: src [N][I][PS] f32 (conv), k = pos*I+ich
// ---------------------------------------------------------------------------
__global__ __launch_bounds__(256)
void pack_w(const float* __restrict__ src, bf16* __restrict__ dst,
            int N, int Kp, int mode, int I, int PS)
{
    int idx = blockIdx.x * 256 + threadIdx.x;
    if (idx >= Kp * N) return;
    int k = idx / N, n = idx - k * N;
    int kb = k >> 5, kk = k & 31;
    float v = 0.0f;
    if (mode == 0) {
        int Ksrc = I * PS;
        if (k < Ksrc) v = src[(size_t)k * N + n];
    } else {
        int pos = k / I, ich = k - pos * I;
        if (pos < PS) v = src[((size_t)n * I + ich) * PS + pos];
    }
    dst[((size_t)kb * N + n) * 32 + kk] = (bf16)v;
}

// ---------------------------------------------------------------------------
// conv1 im2col: x f32 (512,1,28,28) -> A1 bf16 [204800][96] (K=81 zero-padded)
// ---------------------------------------------------------------------------
__global__ __launch_bounds__(256)
void im2col_conv1(const float* __restrict__ x, bf16* __restrict__ A, int total)
{
    int idx = blockIdx.x * 256 + threadIdx.x;
    if (idx >= total) return;
    int m = idx / 96, k = idx - m * 96;
    float v = 0.0f;
    if (k < 81) {
        int b = m / 400, rem = m - b * 400;
        int yy = rem / 20, xx = rem - yy * 20;
        int r = k / 9, s = k - r * 9;
        v = x[(size_t)b * 784 + (yy + r) * 28 + (xx + s)];
    }
    A[idx] = (bf16)v;
}

// ---------------------------------------------------------------------------
// bf16 WMMA GEMM, 32x(16*NT) macro-tile per wave (MT=2 row-tiles x NT col-tiles)
// AMODE 0: A explicit [M][lda].  AMODE 1: implicit NHWC conv, k = pos*I+ch
// ACT: 0 none, 1 relu, 2 sigmoid.  OUTBF: bf16 vs f32 out.
// 8*NT/4 ... per k-step: 2*MT A chunks + 2*NT B chunks feed MT*NT WMMAs.
// ---------------------------------------------------------------------------
template <int AMODE, int ACT, int OUTBF, int NT>
__global__ __launch_bounds__(256)
void gemm_wmma(const bf16* __restrict__ A, int lda,
               const bf16* __restrict__ Bp,
               const float* __restrict__ bias, float scale,
               void* __restrict__ Cout,
               int M, int N, int Kp, Geom g)
{
    const int lane  = threadIdx.x & 31;
    const int wave  = blockIdx.x * 8 + (threadIdx.x >> 5);
    const int tilesN = N / (16 * NT);
    const int tm = wave / tilesN, tn = wave - tm * tilesN;
    const int m0 = tm * 32, n0 = tn * 16 * NT;
    if (m0 >= M) return;

    const int arow = lane & 15;               // A row / B column within tile
    const int kb8  = (lane >> 4) << 3;        // A K-chunk base (0 or 8)
    const int kb16 = (lane >> 4) << 3;        // B K-chunk base (0 or 8)

    v8f acc[2][NT];
    #pragma unroll
    for (int mt = 0; mt < 2; ++mt)
        #pragma unroll
        for (int nt = 0; nt < NT; ++nt) acc[mt][nt] = (v8f){};

    // Per-row-tile A addressing state
    const bf16* aptr[2] = {nullptr, nullptr};
    int b[2], ii[2], iy0[2], ix0[2];
    #pragma unroll
    for (int mt = 0; mt < 2; ++mt) {
        int m = m0 + mt * 16 + arow;
        if (AMODE == 0) {
            aptr[mt] = A + (size_t)m * lda;
        } else {
            int b2 = m / g.OHOW;
            int sp = m - b2 * g.OHOW;
            int py = sp / g.OW, px = sp - py * g.OW;
            b[mt]  = b2 / g.G;  ii[mt] = b2 - b[mt] * g.G;
            iy0[mt] = py * g.stride;  ix0[mt] = px * g.stride;
        }
    }

    for (int k0 = 0; k0 < Kp; k0 += 32) {
        FragU fa[2], fb[NT];
        #pragma unroll
        for (int mt = 0; mt < 2; ++mt) {
            if (AMODE == 0) {
                fa[mt].q[0] = *(const uint4*)(aptr[mt] + k0 + kb8);
                fa[mt].q[1] = *(const uint4*)(aptr[mt] + k0 + 16 + kb8);
            } else {
                #pragma unroll
                for (int c = 0; c < 2; ++c) {
                    int kc  = k0 + kb8 + c * 16;
                    int pos = kc >> g.iShift;
                    int at0 = kc - (pos << g.iShift);
                    uint4 q = {0u, 0u, 0u, 0u};
                    if (pos < g.PS) {
                        int r = pos / g.S, s = pos - r * g.S;
                        size_t off = ((size_t)((b[mt] * g.H + iy0[mt] + r) * g.W
                                     + ix0[mt] + s)) * g.C
                                   + (ii[mt] << g.iShift) + at0;
                        q = *(const uint4*)(A + off);
                    }
                    fa[mt].q[c] = q;
                }
            }
        }
        const bf16* bbase = Bp + ((size_t)(k0 >> 5) * N + (n0 + arow)) * 32 + kb16;
        #pragma unroll
        for (int nt = 0; nt < NT; ++nt) {
            const bf16* bp = bbase + (size_t)nt * 16 * 32;
            fb[nt].q[0] = *(const uint4*)(bp);
            fb[nt].q[1] = *(const uint4*)(bp + 16);
        }
        if (k0 + 32 < Kp)
            __builtin_prefetch(bbase + (size_t)N * 32, 0, 3);  // next B k-block, near cache

        #pragma unroll
        for (int nt = 0; nt < NT; ++nt)
            #pragma unroll
            for (int mt = 0; mt < 2; ++mt)
                acc[mt][nt] = __builtin_amdgcn_wmma_f32_16x16x32_bf16(
                                  false, fa[mt].v, false, fb[nt].v,
                                  (short)0, acc[mt][nt], false, false);
    }

    // C layout: VGPR r, lanes 0-15 -> M=base+r, lanes 16-31 -> M=base+8+r
    #pragma unroll
    for (int mt = 0; mt < 2; ++mt) {
        const int mrow0 = m0 + mt * 16 + ((lane >> 4) << 3);
        #pragma unroll
        for (int nt = 0; nt < NT; ++nt) {
            const int col  = n0 + nt * 16 + (lane & 15);
            const float bv = bias ? bias[col] : 0.0f;
            #pragma unroll
            for (int r = 0; r < 8; ++r) {
                float v = acc[mt][nt][r] * scale + bv;
                if (ACT == 1)      v = v > 0.0f ? v : 0.0f;
                else if (ACT == 2) v = 1.0f / (1.0f + __expf(-v));
                size_t o = (size_t)(mrow0 + r) * N + col;
                if (OUTBF) ((bf16*)Cout)[o] = (bf16)v;
                else       ((float*)Cout)[o] = v;
            }
        }
    }
}

// ---------------------------------------------------------------------------
// Primary caps (num_routing==1): preact = votes/32 + bias (from GEMM epilogue);
// squash over groups of 8 channels; f32 [32768][256] -> bf16 NHWC [32768][256]
// ---------------------------------------------------------------------------
__global__ __launch_bounds__(256)
void prim_squash(const float* __restrict__ pre, bf16* __restrict__ p, int total)
{
    int t = blockIdx.x * 256 + threadIdx.x;
    if (t >= total) return;
    int m = t >> 5, grp = t & 31;
    const float* q = pre + (size_t)m * 256 + grp * 8;
    float n2 = 0.0f;
    #pragma unroll
    for (int a = 0; a < 8; ++a) { float xv = q[a]; n2 += xv * xv; }
    float f = (n2 / (1.0f + n2)) * rsqrtf(n2 + 1e-8f);
    bf16* o = p + (size_t)m * 256 + grp * 8;
    #pragma unroll
    for (int a = 0; a < 8; ++a) o[a] = (bf16)(q[a] * f);
}

// ---------------------------------------------------------------------------
// Conv-caps dynamic routing (3 iters): one block per (b, spatial) site.
// ---------------------------------------------------------------------------
__global__ __launch_bounds__(256)
void route_conv(const bf16* __restrict__ votes, const float* __restrict__ bias,
                bf16* __restrict__ cOut)
{
    __shared__ float v[32 * 256];
    __shared__ float lg[1024], route[1024];
    __shared__ float pre[256], actS[256];
    const int tid  = threadIdx.x;
    const int site = blockIdx.x;           // b*4 + sp
    const int b    = site >> 2, sp = site & 3;

    for (int idx = tid; idx < 32 * 256; idx += 256) {
        int i = idx >> 8, n = idx & 255;
        v[idx] = (float)votes[((size_t)(b * 32 + i) * 4 + sp) * 256 + n];
    }
    for (int idx = tid; idx < 1024; idx += 256) lg[idx] = 0.0f;
    __syncthreads();

    for (int it = 0; it < 3; ++it) {
        if (tid < 32) {                                  // softmax over od per i
            float mx = -1e30f;
            for (int od = 0; od < 32; ++od) mx = fmaxf(mx, lg[tid * 32 + od]);
            float s = 0.0f;
            for (int od = 0; od < 32; ++od) {
                float e = __expf(lg[tid * 32 + od] - mx);
                route[tid * 32 + od] = e; s += e;
            }
            float inv = 1.0f / s;
            for (int od = 0; od < 32; ++od) route[tid * 32 + od] *= inv;
        }
        __syncthreads();
        {                                                // preact (n = tid)
            int od = tid >> 3;
            float s = bias[tid];
            for (int i = 0; i < 32; ++i) s += route[i * 32 + od] * v[i * 256 + tid];
            pre[tid] = s;
        }
        __syncthreads();
        {                                                // squash over 8 atoms
            int gb = (tid >> 3) << 3;
            float n2 = 0.0f;
            for (int a = 0; a < 8; ++a) { float t = pre[gb + a]; n2 += t * t; }
            actS[tid] = pre[tid] * (n2 / (1.0f + n2)) * rsqrtf(n2 + 1e-8f);
        }
        __syncthreads();
        if (it < 2) {                                    // logits update
            for (int p = tid; p < 1024; p += 256) {
                int i = p >> 5, od = p & 31;
                float s = 0.0f;
                for (int a = 0; a < 8; ++a) s += v[i * 256 + od * 8 + a] * actS[od * 8 + a];
                lg[p] += s;
            }
            __syncthreads();
        }
    }
    cOut[(size_t)site * 256 + tid] = (bf16)actS[tid];
}

// ---------------------------------------------------------------------------
// Digit-caps routing (3 iters): one block per batch sample.
// ---------------------------------------------------------------------------
__global__ __launch_bounds__(256)
void route_digit(const bf16* __restrict__ votes, const float* __restrict__ bias,
                 const float* __restrict__ y,
                 float* __restrict__ dcap, bf16* __restrict__ masked)
{
    __shared__ float v[32 * 160];
    __shared__ float lg[320], route[320];
    __shared__ float pre[160], actS[160];
    const int tid = threadIdx.x;
    const int b   = blockIdx.x;

    for (int idx = tid; idx < 32 * 160; idx += 256)
        v[idx] = (float)votes[(size_t)b * 5120 + idx];
    for (int idx = tid; idx < 320; idx += 256) lg[idx] = 0.0f;
    __syncthreads();

    for (int it = 0; it < 3; ++it) {
        if (tid < 32) {
            float mx = -1e30f;
            for (int od = 0; od < 10; ++od) mx = fmaxf(mx, lg[tid * 10 + od]);
            float s = 0.0f;
            for (int od = 0; od < 10; ++od) {
                float e = __expf(lg[tid * 10 + od] - mx);
                route[tid * 10 + od] = e; s += e;
            }
            float inv = 1.0f / s;
            for (int od = 0; od < 10; ++od) route[tid * 10 + od] *= inv;
        }
        __syncthreads();
        if (tid < 160) {
            int od = tid >> 4;
            float s = bias[tid];
            for (int i = 0; i < 32; ++i) s += route[i * 10 + od] * v[i * 160 + tid];
            pre[tid] = s;
        }
        __syncthreads();
        if (tid < 160) {
            int gb = (tid >> 4) << 4;
            float n2 = 0.0f;
            for (int a = 0; a < 16; ++a) { float t = pre[gb + a]; n2 += t * t; }
            actS[tid] = pre[tid] * (n2 / (1.0f + n2)) * rsqrtf(n2 + 1e-8f);
        }
        __syncthreads();
        if (it < 2) {
            for (int p = tid; p < 320; p += 256) {
                int i = p / 10, od = p - i * 10;
                float s = 0.0f;
                for (int a = 0; a < 16; ++a) s += v[i * 160 + od * 16 + a] * actS[od * 16 + a];
                lg[p] += s;
            }
            __syncthreads();
        }
    }
    if (tid < 160) {
        int od = tid >> 4;
        float a = actS[tid];
        dcap[(size_t)b * 160 + tid]   = a;
        masked[(size_t)b * 160 + tid] = (bf16)(a * y[b * 10 + od]);
    }
}

// ---------------------------------------------------------------------------
extern "C" void kernel_launch(void* const* d_in, const int* in_sizes, int n_in,
                              void* d_out, int out_size, void* d_ws, size_t ws_size,
                              hipStream_t stream)
{
    (void)in_sizes; (void)n_in; (void)out_size; (void)ws_size;
    const float* x       = (const float*)d_in[0];
    const float* y       = (const float*)d_in[1];
    const float* conv1_w = (const float*)d_in[2];
    const float* conv1_b = (const float*)d_in[3];
    const float* prim_w  = (const float*)d_in[4];
    const float* prim_b  = (const float*)d_in[5];
    const float* conv_w  = (const float*)d_in[6];
    const float* conv_b  = (const float*)d_in[7];
    const float* digit_w = (const float*)d_in[8];
    const float* digit_b = (const float*)d_in[9];
    const float* fc1_w   = (const float*)d_in[10];
    const float* fc1_b   = (const float*)d_in[11];
    const float* fc2_w   = (const float*)d_in[12];
    const float* fc2_b   = (const float*)d_in[13];
    const float* fc3_w   = (const float*)d_in[14];
    const float* fc3_b   = (const float*)d_in[15];

    char* ws = (char*)d_ws;
    size_t off = 0;
    auto alloc = [&](size_t bytes) -> char* {
        char* p = ws + off;
        off = (off + bytes + 255) & ~(size_t)255;
        return p;
    };

    bf16* B1p = (bf16*)alloc((size_t)96   * 256  * 2);
    bf16* Bpp = (bf16*)alloc((size_t)6400 * 256  * 2);
    bf16* Bcp = (bf16*)alloc((size_t)224  * 256  * 2);
    bf16* Bdp = (bf16*)alloc((size_t)32   * 160  * 2);
    bf16* Bf1 = (bf16*)alloc((size_t)160  * 512  * 2);
    bf16* Bf2 = (bf16*)alloc((size_t)512  * 1024 * 2);
    bf16* Bf3 = (bf16*)alloc((size_t)1024 * 784  * 2);

    bf16*  A1   = (bf16*) alloc((size_t)204800 * 96  * 2);
    bf16*  h1   = (bf16*) alloc((size_t)204800 * 256 * 2);
    float* preP = (float*)alloc((size_t)32768  * 256 * 4);
    bf16*  pBuf = (bf16*) alloc((size_t)32768  * 256 * 2);
    bf16*  v2   = (bf16*) alloc((size_t)65536  * 256 * 2);
    bf16*  cBuf = (bf16*) alloc((size_t)2048   * 256 * 2);
    bf16*  v3   = (bf16*) alloc((size_t)16384  * 160 * 2);
    bf16*  mskd = (bf16*) alloc((size_t)512    * 160 * 2);
    bf16*  r1   = (bf16*) alloc((size_t)512    * 512 * 2);
    bf16*  r2   = (bf16*) alloc((size_t)512    * 1024* 2);

    float* dcap  = (float*)d_out;
    float* recon = (float*)d_out + 81920;

    Geom gz = {};
    Geom gp = {20, 20, 256, 8, 25, 5, 2, 8, 64, 1};   // prim caps conv
    Geom gc = {8, 8, 256, 3, 25, 5, 2, 2, 4, 32};     // conv caps
    Geom gd = {2, 2, 256, 3, 4, 2, 1, 1, 1, 32};      // digit caps

    // ---- weight packing -------------------------------------------------
    pack_w<<<(96   * 256  + 255) / 256, 256, 0, stream>>>(conv1_w, B1p, 256,  96,   1, 1,    81);
    pack_w<<<(6400 * 256  + 255) / 256, 256, 0, stream>>>(prim_w,  Bpp, 256,  6400, 1, 256,  25);
    pack_w<<<(224  * 256  + 255) / 256, 256, 0, stream>>>(conv_w,  Bcp, 256,  224,  1, 8,    25);
    pack_w<<<(32   * 160  + 255) / 256, 256, 0, stream>>>(digit_w, Bdp, 160,  32,   1, 8,    4);
    pack_w<<<(160  * 512  + 255) / 256, 256, 0, stream>>>(fc1_w,   Bf1, 512,  160,  0, 160,  1);
    pack_w<<<(512  * 1024 + 255) / 256, 256, 0, stream>>>(fc2_w,   Bf2, 1024, 512,  0, 512,  1);
    pack_w<<<(1024 * 784  + 255) / 256, 256, 0, stream>>>(fc3_w,   Bf3, 784,  1024, 0, 1024, 1);

    // ---- conv1: im2col + GEMM (relu, bf16 NHWC out) ---------------------
    // tiles: (204800/32) * (256/64) = 6400*4 = 25600 waves -> 3200 blocks
    im2col_conv1<<<(204800 * 96 + 255) / 256, 256, 0, stream>>>(x, A1, 204800 * 96);
    gemm_wmma<0, 1, 1, 4><<<3200, 256, 0, stream>>>(A1, 96, B1p, conv1_b, 1.0f,
                                                    h1, 204800, 256, 96, gz);

    // ---- primary caps: implicit-GEMM (scale 1/32 + bias) then squash ----
    // (32768/32)*(256/64) = 1024*4 = 4096 waves -> 512 blocks
    gemm_wmma<1, 0, 0, 4><<<512, 256, 0, stream>>>(h1, 0, Bpp, prim_b, 1.0f / 32.0f,
                                                   preP, 32768, 256, 6400, gp);
    prim_squash<<<(32768 * 32) / 256, 256, 0, stream>>>(preP, pBuf, 32768 * 32);

    // ---- conv caps: votes (implicit-GEMM) + 3-iter routing --------------
    // (65536/32)*(256/64) = 2048*4 = 8192 waves -> 1024 blocks
    gemm_wmma<1, 0, 1, 4><<<1024, 256, 0, stream>>>(pBuf, 0, Bcp, nullptr, 1.0f,
                                                    v2, 65536, 256, 224, gc);
    route_conv<<<2048, 256, 0, stream>>>(v2, conv_b, cBuf);

    // ---- digit caps: votes (implicit-GEMM) + 3-iter routing -------------
    // (16384/32)*(160/32) = 512*5 = 2560 waves -> 320 blocks
    gemm_wmma<1, 0, 1, 2><<<320, 256, 0, stream>>>(cBuf, 0, Bdp, nullptr, 1.0f,
                                                   v3, 16384, 160, 32, gd);
    route_digit<<<512, 256, 0, stream>>>(v3, digit_b, y, dcap, mskd);

    // ---- reconstruction MLP --------------------------------------------
    gemm_wmma<0, 1, 1, 4><<<16, 256, 0, stream>>>(mskd, 160, Bf1, fc1_b, 1.0f,
                                                  r1, 512, 512, 160, gz);
    gemm_wmma<0, 1, 1, 4><<<32, 256, 0, stream>>>(r1, 512, Bf2, fc2_b, 1.0f,
                                                  r2, 512, 1024, 512, gz);
    // (512/32)*(784/16) = 16*49 = 784 waves -> 98 blocks
    gemm_wmma<0, 2, 0, 1><<<98, 256, 0, stream>>>(r2, 1024, Bf3, fc3_b, 1.0f,
                                                  recon, 512, 784, 1024, gz);
}
